// DinoEncoderModule_7808250544438
// MI455X (gfx1250) — matrix operand order repaired and checked by reading, not compiled, not run on hardware
//
#include <hip/hip_runtime.h>
#include <hip/hip_bf16.h>
#include <math.h>

// ---------------- problem constants ----------------
#define BB 128
#define NN 80
#define ATOMC 16
#define HG 512
#define ENC_BLOCKS 4
#define ENC_HEADS 8
#define N_RBF 16
#define HID 256
#define DEC_LAYERS 8
#define DEC_HEADS 8
#define KLD 6
#define NEGV (-1.0e9f)

typedef __attribute__((ext_vector_type(16))) _Float16 v16h;
typedef __attribute__((ext_vector_type(8)))  float    v8f;

__device__ inline v8f wmma16(v16h a, v16h b, v8f c) {
  return __builtin_amdgcn_wmma_f32_16x16x32_f16(
      /*neg_a=*/false, a, /*neg_b=*/false, b,
      /*c_mod=*/(short)0, c, /*reuse_a=*/false, /*reuse_b=*/false);
}

// A-fragment K index (16x32 fp16, ISA 7.12.2): lanes 0-15 cover K {0..7,16..23}, lanes 16-31 {8..15,24..31}
__device__ inline int kidxA(int lane, int e) {
  return ((lane >> 4) << 3) + (e & 7) + ((e >> 3) << 4);
}
// B-fragment K index (32x16 fp16): lanes 0-15 cover K {0..7,16..23}, lanes 16-31 {8..15,24..31}; column = lane&15
__device__ inline int kidxB(int lane, int e) {
  return ((lane >> 4) << 3) + (e & 7) + ((e >> 3) << 4);
}

// padded N-group count: GEMM block is 128 columns = 8 groups
__device__ __host__ inline int ngp_of(int N) { return ((N + 127) >> 7) << 3; }

// ======================================================================
// Weight packing: W[K,N] fp32 (row major) -> fp16 fragment layout, padded
// to NGP = ceil(N/128)*8 column groups (zero-filled past N).
//   Wp[((kt*NGP + ng)*32 + lane)*16 + e] = W[kt*32 + ((lane>>4)<<3)+(e&7)+((e>>3)<<4)][ng*16 + (lane&15)]
// K multiple of 32.
// ======================================================================
__global__ __launch_bounds__(256)
void pack_w_kernel(const float* __restrict__ W, _Float16* __restrict__ Wp,
                   int K, int N)
{
  const int NGP = ngp_of(N);
  const size_t total = (size_t)(K >> 5) * NGP * 512;
  size_t i = (size_t)blockIdx.x * 256 + threadIdx.x;
  if (i >= total) return;
  int e    = (int)(i & 15);
  int lane = (int)((i >> 4) & 31);
  size_t f = i >> 9;                 // kt*NGP + ng
  int ng = (int)(f % NGP);
  int kt = (int)(f / NGP);
  int k = kt * 32 + ((lane >> 4) << 3) + (e & 7) + ((e >> 3) << 4);
  int n = ng * 16 + (lane & 15);
  float v = (n < N) ? W[(size_t)k * N + n] : 0.0f;
  Wp[i] = (_Float16)v;
}

// ======================================================================
// WMMA GEMM:  out[M,N] = epilogue(A[M,K] @ W[K,N]) with packed fp16 W.
// Block: 256 threads (8 waves), tile 64(M) x 128(N); wave tile 16x64.
// A staged in LDS in fragment layout; B fragments read directly from Wp
// (unconditional: Wp padded to 128-column blocks).
// epilogue: +bias, act (0=none, 1=silu), +res, *rowscale
// M multiple of 64, K multiple of 32. N arbitrary.
// ======================================================================
__global__ __launch_bounds__(256)
void gemm_kernel(const float* __restrict__ A, const _Float16* __restrict__ Wp,
                 const float* __restrict__ bias, const float* __restrict__ res,
                 const float* __restrict__ rowscale, float* __restrict__ out,
                 int M, int N, int K, int act)
{
  __shared__ __attribute__((aligned(32))) _Float16 Asf[4][32][16]; // 4KB, fragment layout
  const int tid  = threadIdx.x;
  const int lane = tid & 31;
  const int wave = tid >> 5;
  const int m0 = blockIdx.y * 64;
  const int n0 = blockIdx.x * 128;
  const int wrow = wave >> 1;   // 0..3
  const int wcol = wave & 1;    // 0..1
  const int NGP = ngp_of(N);

  v8f acc0 = {}, acc1 = {}, acc2 = {}, acc3 = {};

  // per-wave base N-group of the B fragments in Wp
  const int ngbase = (n0 >> 4) + wcol * 4;

  for (int k0 = 0; k0 < K; k0 += 32) {
    // stage A tile 64x32 fp32 -> fragment-layout fp16 (512 float4, 2/thread)
#pragma unroll
    for (int j = 0; j < 2; ++j) {
      int c  = tid + j * 256;
      int r  = c >> 3;             // row 0..63
      int k4 = (c & 7) << 2;       // 0,4,...,28
      const float4 v = *(const float4*)(A + (size_t)(m0 + r) * K + k0 + k4);
      int t  = r >> 4;
      int m  = r & 15;
      int dl = m + (((k4 >> 3) & 1) << 4);
      int e0 = (k4 & 7) | ((k4 >> 4) << 3);
      _Float16* d = &Asf[t][dl][e0];
      d[0] = (_Float16)v.x; d[1] = (_Float16)v.y;
      d[2] = (_Float16)v.z; d[3] = (_Float16)v.w;
    }
    __syncthreads();

    v16h a = *(const v16h*)(&Asf[wrow][lane][0]);
    const _Float16* bp =
        Wp + (((size_t)(k0 >> 5) * NGP + ngbase) * 32 + lane) * 16;
    v16h b0 = *(const v16h*)(bp);
    v16h b1 = *(const v16h*)(bp + 512);
    v16h b2 = *(const v16h*)(bp + 1024);
    v16h b3 = *(const v16h*)(bp + 1536);
    acc0 = wmma16(a, b0, acc0);
    acc1 = wmma16(a, b1, acc1);
    acc2 = wmma16(a, b2, acc2);
    acc3 = wmma16(a, b3, acc3);
    __syncthreads();
  }

  // epilogue: C/D layout — lane&15 = N, vgpr v => M = v + 8*(lane>=16)
  const int mofs = (lane >> 4) << 3;
  const int nl   = lane & 15;
#pragma unroll
  for (int g = 0; g < 4; ++g) {
    v8f acc = (g == 0) ? acc0 : (g == 1) ? acc1 : (g == 2) ? acc2 : acc3;
#pragma unroll
    for (int v = 0; v < 8; ++v) {
      int m = m0 + wrow * 16 + mofs + v;
      int n = n0 + wcol * 64 + g * 16 + nl;
      if (n < N) {
        float val = acc[v];
        if (bias) val += bias[n];
        if (act == 1) val = val / (1.0f + __expf(-val));
        if (res)  val += res[(size_t)m * N + n];
        if (rowscale) val *= rowscale[m];
        out[(size_t)m * N + n] = val;
      }
    }
  }
}

// ======================================================================
// Fused attention: one block per (b, head, 16-query-row block).
// MODE 0: encoder (RBF distance bias + padding/self mask)
// MODE 1: decoder (padding mask, S = 2N, mask index col % Nmask)
// ======================================================================
template<int S, int DH, int MODE>
__global__ __launch_bounds__(256)
void attn_kernel(const float* __restrict__ qkv, int ldq, int koff, int voff,
                 const float* __restrict__ dist, const float* __restrict__ rbfw,
                 const int* __restrict__ pmask, float* __restrict__ o,
                 int H, int Nmask)
{
  constexpr int SP = (S + 31) & ~31;
  __shared__ __attribute__((aligned(16))) float sl[16][SP];

  const int nqb  = S / 16;
  const int bid  = blockIdx.x;
  const int qblk = bid % nqb;
  const int h    = (bid / nqb) % H;
  const int b    = bid / (nqb * H);
  const int tid  = threadIdx.x;
  const int lane = tid & 31;
  const int wave = tid >> 5;
  const int q0   = qblk * 16;
  const int mloc = lane & 15;

  for (int i = tid; i < 16 * SP; i += 256) ((float*)sl)[i] = 0.0f;
  __syncthreads();

  const float scale = rsqrtf((float)DH);

  // ---- logits = Q K^T * scale + bias, into sl ----
  for (int nt = wave; nt < S / 16; nt += 8) {
    v8f acc = {};
#pragma unroll
    for (int ks = 0; ks < DH; ks += 32) {
      const float* qrow = qkv + ((size_t)b * S + q0 + mloc) * ldq + h * DH + ks;
      const float* krow = qkv + ((size_t)b * S + nt * 16 + mloc) * ldq + koff + h * DH + ks;
      v16h a, bf;
#pragma unroll
      for (int e = 0; e < 16; ++e) {
        a[e]  = (_Float16)qrow[kidxA(lane, e)];
        bf[e] = (_Float16)krow[kidxB(lane, e)];
      }
      acc = wmma16(a, bf, acc);
    }
    const int mofs = (lane >> 4) << 3;
#pragma unroll
    for (int v = 0; v < 8; ++v) {
      int qr  = mofs + v;              // local query row
      int col = nt * 16 + mloc;        // key index
      float val = acc[v] * scale;
      if (MODE == 0) {
        int qg = q0 + qr;
        bool msk = (pmask[b * Nmask + col] != 0) || (qg == col);
        if (msk) {
          val += NEGV;
        } else {
          float d  = dist[((size_t)b * S + qg) * S + col];
          float db = 0.0f;
#pragma unroll
          for (int r = 0; r < N_RBF; ++r) {
            float cr = (float)r * (10.0f / 15.0f);
            float t  = d - cr;
            db += __expf(-2.0f * t * t) * rbfw[r * H + h];
          }
          val += db;
        }
      } else {
        if (pmask[b * Nmask + (col % Nmask)] != 0) val += NEGV;
      }
      sl[qr][col] = val;
    }
  }
  __syncthreads();

  // ---- softmax over key dim (2 rows per wave) ----
  for (int r = wave; r < 16; r += 8) {
    float mx = -3.4e38f;
    for (int c = lane; c < S; c += 32) mx = fmaxf(mx, sl[r][c]);
#pragma unroll
    for (int off = 16; off; off >>= 1) mx = fmaxf(mx, __shfl_xor(mx, off, 32));
    float sum = 0.0f;
    for (int c = lane; c < S; c += 32) {
      float e = __expf(sl[r][c] - mx);
      sl[r][c] = e;
      sum += e;
    }
#pragma unroll
    for (int off = 16; off; off >>= 1) sum += __shfl_xor(sum, off, 32);
    float inv = 1.0f / (sum + 1e-20f);
    for (int c = lane; c < S; c += 32) sl[r][c] *= inv;
  }
  __syncthreads();

  // ---- out = attn @ V ----
  for (int nt = wave; nt < DH / 16; nt += 8) {
    v8f acc = {};
    for (int ks = 0; ks < SP; ks += 32) {
      v16h a, bf;
#pragma unroll
      for (int e = 0; e < 16; ++e) {
        int ka = ks + kidxA(lane, e);
        int kb = ks + kidxB(lane, e);
        a[e] = (_Float16)sl[mloc][ka];
        float vv = 0.0f;
        if (kb < S)
          vv = qkv[((size_t)b * S + kb) * ldq + voff + h * DH + nt * 16 + mloc];
        bf[e] = (_Float16)vv;
      }
      acc = wmma16(a, bf, acc);
    }
    const int mofs = (lane >> 4) << 3;
#pragma unroll
    for (int v = 0; v < 8; ++v) {
      int qr = mofs + v;
      o[((size_t)b * S + q0 + qr) * (H * DH) + h * DH + nt * 16 + mloc] = acc[v];
    }
  }
}

// ======================================================================
// LayerNorm: one block (128 threads) per row
// ======================================================================
__global__ __launch_bounds__(128)
void ln_kernel(const float* __restrict__ x, const float* __restrict__ g,
               const float* __restrict__ bta, float* __restrict__ y, int D)
{
  const int row = blockIdx.x;
  const float* xr = x + (size_t)row * D;
  float s = 0.0f, s2 = 0.0f;
  for (int c = threadIdx.x; c < D; c += 128) {
    float v = xr[c];
    s += v; s2 += v * v;
  }
  __shared__ float sh[2][4];
#pragma unroll
  for (int off = 16; off; off >>= 1) {
    s  += __shfl_xor(s, off, 32);
    s2 += __shfl_xor(s2, off, 32);
  }
  if ((threadIdx.x & 31) == 0) {
    sh[0][threadIdx.x >> 5] = s;
    sh[1][threadIdx.x >> 5] = s2;
  }
  __syncthreads();
  s  = sh[0][0] + sh[0][1] + sh[0][2] + sh[0][3];
  s2 = sh[1][0] + sh[1][1] + sh[1][2] + sh[1][3];
  float mu  = s / (float)D;
  float var = s2 / (float)D - mu * mu;
  float inv = rsqrtf(var + 1e-5f);
  for (int c = threadIdx.x; c < D; c += 128)
    y[(size_t)row * D + c] = (xr[c] - mu) * inv * g[c] + bta[c];
}

// ======================================================================
// small helper kernels
// ======================================================================
__global__ void zero_kernel(float* p, int n) {
  int i = blockIdx.x * blockDim.x + threadIdx.x;
  if (i < n) p[i] = 0.0f;
}

__global__ void real_kernel(const int* __restrict__ pmask, float* __restrict__ rl, int n) {
  int i = blockIdx.x * blockDim.x + threadIdx.x;
  if (i < n) rl[i] = (pmask[i] != 0) ? 0.0f : 1.0f;
}

__global__ void dist_kernel(const float* __restrict__ coord, float* __restrict__ dist, int n) {
  int i = blockIdx.x * blockDim.x + threadIdx.x;
  if (i >= n) return;
  int b   = i / (NN * NN);
  int rem = i % (NN * NN);
  int q = rem / NN, k = rem % NN;
  const float* cq = coord + ((size_t)b * NN + q) * 3;
  const float* ck = coord + ((size_t)b * NN + k) * 3;
  float dx = cq[0] - ck[0], dy = cq[1] - ck[1], dz = cq[2] - ck[2];
  dist[i] = sqrtf(dx * dx + dy * dy + dz * dz + 1e-12f);
}

__global__ void embed_kernel(const float* __restrict__ onehot, const float* __restrict__ table,
                             float* __restrict__ out, int rows, int D) {
  int i = blockIdx.x * blockDim.x + threadIdx.x;
  if (i >= rows * D) return;
  int r = i / D, c = i % D;
  const float* a = onehot + (size_t)r * ATOMC;
  int am = 0; float best = a[0];
#pragma unroll
  for (int j = 1; j < ATOMC; ++j) if (a[j] > best) { best = a[j]; am = j; }
  out[i] = table[am * D + c];
}

__global__ __launch_bounds__(64)
void quant_kernel(const float* __restrict__ rep, const float* __restrict__ qw,
                  const float* __restrict__ qb, const float* __restrict__ rl,
                  float* __restrict__ z, float* __restrict__ klsum)
{
  const int row = blockIdx.x;
  __shared__ float mo[2 * KLD];
  int t = threadIdx.x;
  if (t < 2 * KLD) {
    float s = qb[t];
    const float* r = rep + (size_t)row * HID;
    for (int k = 0; k < HID; ++k) s += r[k] * qw[k * (2 * KLD) + t];
    mo[t] = s;
    if (t < KLD) z[row * KLD + t] = s;   // z = mu (eval)
  }
  __syncthreads();
  if (t == 0) {
    float kl = 0.0f;
#pragma unroll
    for (int j = 0; j < KLD; ++j) {
      float mu = mo[j], lv = mo[KLD + j];
      kl += (1.0f + lv - mu * mu - __expf(lv));
    }
    atomicAdd(klsum, kl * rl[row]);
  }
}

__global__ __launch_bounds__(256)
void dec_input_kernel(const float* __restrict__ coord_t, const float* __restrict__ atomics_t,
                      const float* __restrict__ tvec, const float* __restrict__ z,
                      const float* __restrict__ qout_w, const float* __restrict__ qout_b,
                      const float* __restrict__ cond, const float* __restrict__ lin_w,
                      const float* __restrict__ atom_emb, const float* __restrict__ rl,
                      float* __restrict__ x)
{
  const int row = blockIdx.x;   // 0..B*2N
  const int c   = threadIdx.x;  // 0..HID-1
  const int b = row / (2 * NN);
  const int s = row % (2 * NN);
  const bool second = (s >= NN);
  const int n = second ? s - NN : s;
  int i2 = c & ~1;
  float div = __expf(-logf(10000.0f) * (float)i2 / (float)HID);
  float ang = (float)n * div;
  float pe = (c & 1) ? __cosf(ang) : __sinf(ang);
  float r = rl[b * NN + n];
  float val;
  if (!second) {
    const float* co = coord_t + ((size_t)b * NN + n) * 3;
    float lin = co[0] * lin_w[0 * HID + c] + co[1] * lin_w[1 * HID + c] + co[2] * lin_w[2 * HID + c];
    const float* ao = atomics_t + ((size_t)b * NN + n) * ATOMC;
    int am = 0; float best = ao[0];
#pragma unroll
    for (int j = 1; j < ATOMC; ++j) if (ao[j] > best) { best = ao[j]; am = j; }
    float ae = atom_emb[am * HID + c];
    const int half = HID / 2;
    float tb = tvec[b];
    float tf;
    if (c < half) {
      float f = __expf(-logf(200.0f) * (float)c / (float)half);
      tf = __sinf(tb * 200.0f * f);
    } else {
      float f = __expf(-logf(200.0f) * (float)(c - half) / (float)half);
      tf = __cosf(tb * 200.0f * f);
    }
    val = (lin + ae + pe + tf) * r + cond[0 * HID + c];
  } else {
    const float* zr = z + ((size_t)b * NN + n) * KLD;
    float e = qout_b[c];
#pragma unroll
    for (int j = 0; j < KLD; ++j) e += zr[j] * qout_w[j * HID + c];
    val = (e + pe) * r + cond[1 * HID + c];
  }
  x[(size_t)row * HID + c] = val;
}

__global__ void hout_kernel(const float* __restrict__ x, const float* __restrict__ rl,
                            float* __restrict__ hout, int n) {
  int i = blockIdx.x * blockDim.x + threadIdx.x;
  if (i >= n) return;
  int r = i / HID, c = i % HID;
  int b = r / NN, nn = r % NN;
  hout[i] = x[((size_t)b * 2 * NN + nn) * HID + c] * rl[r];
}

__global__ void coords_kernel(const float* __restrict__ lnh, const float* __restrict__ ocw,
                              float* __restrict__ out, int rows) {
  int i = blockIdx.x * blockDim.x + threadIdx.x;
  if (i >= rows * 3) return;
  int r = i / 3, c = i % 3;
  const float* x = lnh + (size_t)r * HID;
  float s = 0.0f;
  for (int k = 0; k < HID; ++k) s += x[k] * ocw[k * 3 + c];
  out[i] = s;
}

__global__ void kl_final_kernel(const float* __restrict__ klsum, float* __restrict__ out) {
  out[0] = -0.5f * (klsum[0] / (float)(BB * NN)) * 1e-6f;
}

// ======================================================================
// host launcher
// ======================================================================
extern "C" void kernel_launch(void* const* d_in, const int* in_sizes, int n_in,
                              void* d_out, int out_size, void* d_ws, size_t ws_size,
                              hipStream_t stream) {
  (void)in_sizes; (void)n_in; (void)out_size; (void)ws_size;
  const float* coord_ori  = (const float*)d_in[0];
  const float* atomics_o  = (const float*)d_in[1];
  const int*   pmask      = (const int*)  d_in[2];
  const float* coord_t    = (const float*)d_in[3];
  const float* atomics_t  = (const float*)d_in[4];
  const float* tvec       = (const float*)d_in[5];
  const float* enc_embed_w= (const float*)d_in[6];
  const float* rbf_w      = (const float*)d_in[7];
  const float* e_qkv_w    = (const float*)d_in[8];
  const float* e_qkv_b    = (const float*)d_in[9];
  const float* e_out_w    = (const float*)d_in[10];
  const float* e_out_b    = (const float*)d_in[11];
  const float* e_ff1_w    = (const float*)d_in[12];
  const float* e_ff1_b    = (const float*)d_in[13];
  const float* e_ff2_w    = (const float*)d_in[14];
  const float* e_ff2_b    = (const float*)d_in[15];
  const float* e_ln1_g    = (const float*)d_in[16];
  const float* e_ln1_b    = (const float*)d_in[17];
  const float* e_ln2_g    = (const float*)d_in[18];
  const float* e_ln2_b    = (const float*)d_in[19];
  const float* enc_proj_w = (const float*)d_in[20];
  const float* quant_w    = (const float*)d_in[21];
  const float* quant_b    = (const float*)d_in[22];
  const float* qout_w     = (const float*)d_in[23];
  const float* qout_b     = (const float*)d_in[24];
  const float* cond_embed = (const float*)d_in[25];
  const float* lin_embed_w= (const float*)d_in[26];
  const float* atom_embed = (const float*)d_in[27];
  const float* d_qkv_w    = (const float*)d_in[28];
  const float* d_qkv_b    = (const float*)d_in[29];
  const float* d_out_w    = (const float*)d_in[30];
  const float* d_out_b    = (const float*)d_in[31];
  const float* d_ff1_w    = (const float*)d_in[32];
  const float* d_ff1_b    = (const float*)d_in[33];
  const float* d_ff2_w    = (const float*)d_in[34];
  const float* d_ff2_b    = (const float*)d_in[35];
  const float* d_ln1_g    = (const float*)d_in[36];
  const float* d_ln1_b    = (const float*)d_in[37];
  const float* d_ln2_g    = (const float*)d_in[38];
  const float* d_ln2_b    = (const float*)d_in[39];
  const float* oc_ln_g    = (const float*)d_in[40];
  const float* oc_ln_b    = (const float*)d_in[41];
  const float* oc_w       = (const float*)d_in[42];
  const float* oa1_w      = (const float*)d_in[43];
  const float* oa1_b      = (const float*)d_in[44];
  const float* oa2_w      = (const float*)d_in[45];
  const float* oa2_b      = (const float*)d_in[46];

  float* out = (float*)d_out;

  const int Menc = BB * NN;       // 10240
  const int Mdec = BB * 2 * NN;   // 20480

  // ---- workspace layout (floats; 32B-aligned slots) ----
  float* ws = (float*)d_ws;
  size_t off = 0;
  auto alloc = [&](size_t n) { float* p = ws + off; off += (n + 7) & ~(size_t)7; return p; };
  float* rl    = alloc(Menc);
  float* dist  = alloc((size_t)BB * NN * NN);
  float* z     = alloc((size_t)Menc * KLD);
  float* klsum = alloc(8);
  float* h     = alloc((size_t)Mdec * HID);      // == Menc*HG; reused as decoder x
  float* lnb   = alloc((size_t)Mdec * HID);
  float* qkvb  = alloc((size_t)Mdec * 3 * HID);  // == Menc*3*HG
  float* ob    = alloc((size_t)Mdec * HID);      // attention out; reused as a1
  float* ffb   = alloc((size_t)Mdec * 4 * HID);  // == Menc*4*HG
  float* rep   = alloc((size_t)Menc * HID);      // reused as hout
  _Float16* wpack = (_Float16*)alloc((size_t)1 << 20);  // 2M halves packed-weight scratch

  auto gemm = [&](const float* A, const float* W, const float* bias, const float* res,
                  const float* rowscale, float* O, int M, int Nn, int K, int act) {
    int NGP = ((Nn + 127) >> 7) << 3;
    size_t total = (size_t)(K >> 5) * NGP * 512;
    pack_w_kernel<<<(unsigned)((total + 255) / 256), 256, 0, stream>>>(W, wpack, K, Nn);
    dim3 grid((Nn + 127) / 128, M / 64);
    gemm_kernel<<<grid, 256, 0, stream>>>(A, wpack, bias, res, rowscale, O, M, Nn, K, act);
  };

  zero_kernel<<<1, 32, 0, stream>>>(klsum, 8);
  real_kernel<<<(Menc + 255) / 256, 256, 0, stream>>>(pmask, rl, Menc);
  dist_kernel<<<((BB * NN * NN) + 255) / 256, 256, 0, stream>>>(coord_ori, dist, BB * NN * NN);
  embed_kernel<<<((Menc * HG) + 255) / 256, 256, 0, stream>>>(atomics_o, enc_embed_w, h, Menc, HG);

  // ===== encoder (D=HG=512, heads 8, dh 64, seq 80) =====
  for (int i = 0; i < ENC_BLOCKS; ++i) {
    ln_kernel<<<Menc, 128, 0, stream>>>(h, e_ln1_g + i * HG, e_ln1_b + i * HG, lnb, HG);
    gemm(lnb, e_qkv_w + (size_t)i * HG * 3 * HG, e_qkv_b + (size_t)i * 3 * HG,
         nullptr, nullptr, qkvb, Menc, 3 * HG, HG, 0);
    attn_kernel<NN, 64, 0><<<BB * ENC_HEADS * (NN / 16), 256, 0, stream>>>(
        qkvb, 3 * HG, HG, 2 * HG, dist, rbf_w + (size_t)i * N_RBF * ENC_HEADS, pmask, ob,
        ENC_HEADS, NN);
    gemm(ob, e_out_w + (size_t)i * HG * HG, e_out_b + (size_t)i * HG, h, nullptr, h,
         Menc, HG, HG, 0);
    ln_kernel<<<Menc, 128, 0, stream>>>(h, e_ln2_g + i * HG, e_ln2_b + i * HG, lnb, HG);
    gemm(lnb, e_ff1_w + (size_t)i * HG * 4 * HG, e_ff1_b + (size_t)i * 4 * HG,
         nullptr, nullptr, ffb, Menc, 4 * HG, HG, 1);
    gemm(ffb, e_ff2_w + (size_t)i * 4 * HG * HG, e_ff2_b + (size_t)i * HG, h, nullptr, h,
         Menc, HG, 4 * HG, 0);
  }

  // ===== bottleneck: rep = (h @ enc_proj_w) * real =====
  gemm(h, enc_proj_w, nullptr, nullptr, rl, rep, Menc, HID, HG, 0);
  quant_kernel<<<Menc, 64, 0, stream>>>(rep, quant_w, quant_b, rl, z, klsum);

  // ===== decoder input assembly (x overwrites h buffer) =====
  float* x = h;
  dec_input_kernel<<<Mdec, HID, 0, stream>>>(coord_t, atomics_t, tvec, z, qout_w, qout_b,
                                             cond_embed, lin_embed_w, atom_embed, rl, x);

  // ===== decoder (D=HID=256, heads 8, dh 32, seq 160) =====
  for (int i = 0; i < DEC_LAYERS; ++i) {
    ln_kernel<<<Mdec, 128, 0, stream>>>(x, d_ln1_g + i * HID, d_ln1_b + i * HID, lnb, HID);
    gemm(lnb, d_qkv_w + (size_t)i * HID * 3 * HID, d_qkv_b + (size_t)i * 3 * HID,
         nullptr, nullptr, qkvb, Mdec, 3 * HID, HID, 0);
    attn_kernel<2 * NN, 32, 1><<<BB * DEC_HEADS * (2 * NN / 16), 256, 0, stream>>>(
        qkvb, 3 * HID, HID, 2 * HID, nullptr, nullptr, pmask, ob, DEC_HEADS, NN);
    gemm(ob, d_out_w + (size_t)i * HID * HID, d_out_b + (size_t)i * HID, x, nullptr, x,
         Mdec, HID, HID, 0);
    ln_kernel<<<Mdec, 128, 0, stream>>>(x, d_ln2_g + i * HID, d_ln2_b + i * HID, lnb, HID);
    gemm(lnb, d_ff1_w + (size_t)i * HID * 4 * HID, d_ff1_b + (size_t)i * 4 * HID,
         nullptr, nullptr, ffb, Mdec, 4 * HID, HID, 1);
    gemm(ffb, d_ff2_w + (size_t)i * 4 * HID * HID, d_ff2_b + (size_t)i * HID, x, nullptr, x,
         Mdec, HID, 4 * HID, 0);
  }

  // ===== output heads =====
  float* hout = rep;
  float* a1   = ob;
  hout_kernel<<<((Menc * HID) + 255) / 256, 256, 0, stream>>>(x, rl, hout, Menc * HID);
  ln_kernel<<<Menc, 128, 0, stream>>>(hout, oc_ln_g, oc_ln_b, lnb, HID);
  coords_kernel<<<((Menc * 3) + 255) / 256, 256, 0, stream>>>(lnb, oc_w, out, Menc);
  gemm(hout, oa1_w, oa1_b, nullptr, nullptr, a1, Menc, HID, HID, 1);
  gemm(a1, oa2_w, oa2_b, nullptr, nullptr, out + (size_t)Menc * 3, Menc, ATOMC, HID, 0);
  kl_final_kernel<<<1, 1, 0, stream>>>(klsum, out + (size_t)Menc * 3 + (size_t)Menc * ATOMC);
}
